// ScaledDotProductAttention_30743375904898
// MI455X (gfx1250) — compile-verified, matrix operand order checked by reading, not run on hardware
//
#include <hip/hip_runtime.h>
#include <hip/hip_bf16.h>

// Multi-head attention, MI455X (gfx1250, wave32, WMMA).
// Pipeline: [qkv_proj f32->f16] -> [flash attention, weights*mask streamed once, NT]
//           -> [out proj, NT stores].
// All matmuls use v_wmma_f32_16x16x32_f16 (f32 accumulate).
//
// Cache policy (MI455X: 23.3 TB/s HBM, 192 MB L2):
//  - weights (537MB) + mask (134MB) are single-use streams -> NT loads, so they
//    do not evict the hot Qh/Kh/Vt working set (24MB, re-read ~128x) from L2.
//  - final output (16MB, write-once) -> NT stores.

typedef __attribute__((ext_vector_type(16))) _Float16 v16h;
typedef __attribute__((ext_vector_type(8)))  _Float16 v8h;
typedef __attribute__((ext_vector_type(8)))  float    v8f;

#define BS 2
#define NQ 2048
#define NK 2048
#define DM 1024
#define H  16
#define DH 64

#define WMMA_F32_F16(A, B, C) \
  __builtin_amdgcn_wmma_f32_16x16x32_f16(false, (A), false, (B), (short)0, (C), false, false)

// Pack 8+8 contiguous f32 (two chunks) into a 16xhalf fragment register.
__device__ __forceinline__ v16h pack16_f32(const float* p0, const float* p1) {
  float4 a0 = *(const float4*)p0;
  float4 a1 = *(const float4*)(p0 + 4);
  float4 b0 = *(const float4*)p1;
  float4 b1 = *(const float4*)(p1 + 4);
  v16h r;
  r[0]=(_Float16)a0.x; r[1]=(_Float16)a0.y; r[2]=(_Float16)a0.z; r[3]=(_Float16)a0.w;
  r[4]=(_Float16)a1.x; r[5]=(_Float16)a1.y; r[6]=(_Float16)a1.z; r[7]=(_Float16)a1.w;
  r[8]=(_Float16)b0.x; r[9]=(_Float16)b0.y; r[10]=(_Float16)b0.z; r[11]=(_Float16)b0.w;
  r[12]=(_Float16)b1.x; r[13]=(_Float16)b1.y; r[14]=(_Float16)b1.z; r[15]=(_Float16)b1.w;
  return r;
}

// Pack 8+8 contiguous f16 (two 16B chunks) into a fragment register.
__device__ __forceinline__ v16h pack16_f16(const _Float16* p0, const _Float16* p1) {
  v8h lo = *(const v8h*)p0;
  v8h hi = *(const v8h*)p1;
  v16h r;
#pragma unroll
  for (int i = 0; i < 8; ++i) { r[i] = lo[i]; r[i + 8] = hi[i]; }
  return r;
}

// ---------------------------------------------------------------------------
// Kernel 1: fused Q/K/V projections.  y = X @ W^T + b, cast to f16.
// blockIdx.y selects {Q,K,V}.  One wave -> 16x64 output tile (A reused x4).
// Q,K stored (b,h,n,d); V stored transposed (b,h,d,n) for P.V B-fragments.
// ---------------------------------------------------------------------------
__global__ __launch_bounds__(256) void qkv_proj_kernel(
    const float* __restrict__ queries, const float* __restrict__ keys,
    const float* __restrict__ values,
    const float* __restrict__ Wq, const float* __restrict__ bq,
    const float* __restrict__ Wk, const float* __restrict__ bk,
    const float* __restrict__ Wv, const float* __restrict__ bv,
    _Float16* __restrict__ Qh, _Float16* __restrict__ Kh,
    _Float16* __restrict__ Vt)
{
  const int lane = threadIdx.x & 31;
  const int widx = threadIdx.x >> 5;
  const int wg = blockIdx.x * 8 + widx;   // 4096 waves: 256 row-tiles x 16 col64-tiles
  const int rt = wg >> 4;
  const int ct = wg & 15;
  const int b  = rt >> 7;                 // 128 row-tiles per batch
  const int n0 = (rt & 127) << 4;
  const int o0 = ct << 6;
  const int sel = blockIdx.y;

  const float* X    = (sel == 0) ? queries : (sel == 1) ? keys : values;
  const float* W    = (sel == 0) ? Wq : (sel == 1) ? Wk : Wv;
  const float* bias = (sel == 0) ? bq : (sel == 1) ? bk : bv;

  const int ln = lane & 15;
  const int hw = lane >> 4;
  const int g8 = hw << 3;

  const float* xrow = X + ((size_t)b * NQ + n0 + ln) * DM;

  v8f acc[4] = {};
#pragma unroll 1
  for (int k0 = 0; k0 < DM; k0 += 32) {
    v16h a = pack16_f32(xrow + k0 + g8, xrow + k0 + 16 + g8);
#pragma unroll
    for (int c = 0; c < 4; ++c) {
      const float* wr = W + (size_t)(o0 + 16 * c + ln) * DM + k0 + g8;
      v16h bf = pack16_f32(wr, wr + 16);
      acc[c] = WMMA_F32_F16(a, bf, acc[c]);
    }
  }

#pragma unroll
  for (int c = 0; c < 4; ++c) {
    const int col = o0 + 16 * c + ln;
    const int hh = col >> 6, d = col & 63;
    const float bval = bias[col];
#pragma unroll
    for (int r = 0; r < 8; ++r) {
      const int n = n0 + r + 8 * hw;            // C-layout: row = r + 8*(lane/16)
      const float v = acc[c][r] + bval;
      if (sel == 2) {
        Vt[(((size_t)b * H + hh) * DH + d) * NK + n] = (_Float16)v;
      } else if (sel == 1) {
        Kh[(((size_t)b * H + hh) * NK + n) * DH + d] = (_Float16)v;
      } else {
        Qh[(((size_t)b * H + hh) * NQ + n) * DH + d] = (_Float16)v;
      }
    }
  }
}

// ---------------------------------------------------------------------------
// Kernel 2: flash-style attention.  One wave per (b,h, 16-query tile).
// 32 keys per step: S = Q K^T (4 WMMA), scale * weights (NT), mask (NT) -> -1e30,
// online softmax (16-lane shfl_xor reductions), P transposed to A-fragment
// through per-wave LDS, O += P.V (4 WMMA).  weights+mask (671MB) read once, NT.
// ---------------------------------------------------------------------------
__global__ __launch_bounds__(256) void attn_kernel(
    const _Float16* __restrict__ Qh, const _Float16* __restrict__ Kh,
    const _Float16* __restrict__ Vt,
    const unsigned char* __restrict__ mask, const float* __restrict__ weights,
    _Float16* __restrict__ concat)
{
  __shared__ __align__(16) _Float16 pstage[8][16 * 32];   // 1KB per wave
  const int lane = threadIdx.x & 31;
  const int widx = threadIdx.x >> 5;
  const int wg = blockIdx.x * 8 + widx;   // 4096 waves: (b*H+h)*128 + qtile
  const int qt = wg & 127;
  const int bh = wg >> 7;
  const int h  = bh & (H - 1);
  const int b  = bh >> 4;
  const int q0 = qt << 4;

  const int ln = lane & 15;
  const int hw = lane >> 4;
  const int g8 = hw << 3;

  // Q A-fragments for d=0..31 and d=32..63 (loaded once)
  const _Float16* Qrow = Qh + ((size_t)bh * NQ + q0 + ln) * DH;
  const v16h aQ0 = pack16_f16(Qrow + g8,      Qrow + 16 + g8);
  const v16h aQ1 = pack16_f16(Qrow + 32 + g8, Qrow + 48 + g8);

  const float*         wrow = weights + ((size_t)bh * NQ + q0) * NK;
  const unsigned char* mrow = mask    + ((size_t)bh * NQ + q0) * NK;

  v8f o0f = {}, o1f = {}, o2f = {}, o3f = {};
  float mr[8], lr[8];
#pragma unroll
  for (int r = 0; r < 8; ++r) { mr[r] = -1e30f; lr[r] = 0.0f; }

  _Float16* st = &pstage[widx][0];

#pragma unroll 1
  for (int it = 0; it < NK / 32; ++it) {
    const int kk0 = it * 32;

    // ---- S = Q K^T  (two 16-key column tiles) ----
    v8f s[2] = {};
#pragma unroll
    for (int j = 0; j < 2; ++j) {
      const _Float16* Krow = Kh + ((size_t)bh * NK + kk0 + 16 * j + ln) * DH;
      v16h bK0 = pack16_f16(Krow + g8,      Krow + 16 + g8);
      s[j] = WMMA_F32_F16(aQ0, bK0, s[j]);
      v16h bK1 = pack16_f16(Krow + 32 + g8, Krow + 48 + g8);
      s[j] = WMMA_F32_F16(aQ1, bK1, s[j]);
    }

    // Prefetch next K tile rows (we WANT these resident; weights/mask are NT
    // streams and must not be prefetched into the caches).
    {
      const int kkn = (kk0 + 32 < NK) ? kk0 + 32 : kk0;
      __builtin_prefetch(Kh + ((size_t)bh * NK + kkn + lane) * DH, 0, 3);
    }

    // ---- scale * weights (NT), mask (NT) -> -1e30 ----
    float pv0[8], pv1[8];
#pragma unroll
    for (int r = 0; r < 8; ++r) {
      const size_t off = (size_t)(r + 8 * hw) * NK + kk0 + ln;
      const float w0 = __builtin_nontemporal_load(wrow + off);
      const float w1 = __builtin_nontemporal_load(wrow + off + 16);
      const unsigned char m0 = __builtin_nontemporal_load(mrow + off);
      const unsigned char m1 = __builtin_nontemporal_load(mrow + off + 16);
      float sv0 = s[0][r] * 0.125f * w0;
      float sv1 = s[1][r] * 0.125f * w1;
      if (m0) sv0 = -1e30f;
      if (m1) sv1 = -1e30f;
      pv0[r] = sv0; pv1[r] = sv1;
    }

    // ---- online softmax (rows live across 16 lanes of a half-wave) ----
#pragma unroll
    for (int r = 0; r < 8; ++r) {
      float rm = fmaxf(pv0[r], pv1[r]);
      rm = fmaxf(rm, __shfl_xor(rm, 1, 32));
      rm = fmaxf(rm, __shfl_xor(rm, 2, 32));
      rm = fmaxf(rm, __shfl_xor(rm, 4, 32));
      rm = fmaxf(rm, __shfl_xor(rm, 8, 32));
      const float mnew  = fmaxf(mr[r], rm);
      const float alpha = __expf(mr[r] - mnew);
      const float p0 = __expf(pv0[r] - mnew);
      const float p1 = __expf(pv1[r] - mnew);
      float rs = p0 + p1;
      rs += __shfl_xor(rs, 1, 32);
      rs += __shfl_xor(rs, 2, 32);
      rs += __shfl_xor(rs, 4, 32);
      rs += __shfl_xor(rs, 8, 32);
      lr[r] = lr[r] * alpha + rs;
      mr[r] = mnew;
      pv0[r] = p0; pv1[r] = p1;
      o0f[r] = o0f[r] * alpha;
      o1f[r] = o1f[r] * alpha;
      o2f[r] = o2f[r] * alpha;
      o3f[r] = o3f[r] * alpha;
    }

    // ---- transpose P (C-layout) -> A-fragment layout via per-wave LDS ----
#pragma unroll
    for (int r = 0; r < 8; ++r) {
      const int m = r + 8 * hw;
      st[m * 32 + ln]      = (_Float16)pv0[r];
      st[m * 32 + 16 + ln] = (_Float16)pv1[r];
    }
    asm volatile("s_wait_dscnt 0" ::: "memory");
    v16h aP = pack16_f16(st + ln * 32 + g8, st + ln * 32 + 16 + g8);

    // ---- O += P.V  (V transposed: rows contiguous in key) ----
    {
      const _Float16* v0 = Vt + ((size_t)bh * DH + 0  + ln) * NK + kk0;
      o0f = WMMA_F32_F16(aP, pack16_f16(v0 + g8, v0 + 16 + g8), o0f);
      const _Float16* v1 = Vt + ((size_t)bh * DH + 16 + ln) * NK + kk0;
      o1f = WMMA_F32_F16(aP, pack16_f16(v1 + g8, v1 + 16 + g8), o1f);
      const _Float16* v2 = Vt + ((size_t)bh * DH + 32 + ln) * NK + kk0;
      o2f = WMMA_F32_F16(aP, pack16_f16(v2 + g8, v2 + 16 + g8), o2f);
      const _Float16* v3 = Vt + ((size_t)bh * DH + 48 + ln) * NK + kk0;
      o3f = WMMA_F32_F16(aP, pack16_f16(v3 + g8, v3 + 16 + g8), o3f);
    }
  }

  // ---- normalize and write to concat layout (b, q, h*64 + d) ----
#pragma unroll
  for (int r = 0; r < 8; ++r) lr[r] = 1.0f / lr[r];
#pragma unroll
  for (int r = 0; r < 8; ++r) {
    const int q = q0 + r + 8 * hw;
    _Float16* crow = concat + ((size_t)b * NQ + q) * DM + h * DH + ln;
    crow[0]  = (_Float16)(o0f[r] * lr[r]);
    crow[16] = (_Float16)(o1f[r] * lr[r]);
    crow[32] = (_Float16)(o2f[r] * lr[r]);
    crow[48] = (_Float16)(o3f[r] * lr[r]);
  }
}

// ---------------------------------------------------------------------------
// Kernel 3: output projection.  out = concat @ Wo^T + bo (f32 result, NT stores).
// ---------------------------------------------------------------------------
__global__ __launch_bounds__(256) void out_proj_kernel(
    const _Float16* __restrict__ concat, const float* __restrict__ Wo,
    const float* __restrict__ bo, float* __restrict__ out)
{
  const int lane = threadIdx.x & 31;
  const int widx = threadIdx.x >> 5;
  const int wg = blockIdx.x * 8 + widx;   // 4096 waves: 256 row-tiles x 16 col64
  const int rt = wg >> 4;
  const int ct = wg & 15;
  const int row0 = rt << 4;               // rows over BS*NQ = 4096
  const int o0 = ct << 6;

  const int ln = lane & 15;
  const int hw = lane >> 4;
  const int g8 = hw << 3;

  const _Float16* arow = concat + (size_t)(row0 + ln) * DM;

  v8f acc[4] = {};
#pragma unroll 1
  for (int k0 = 0; k0 < DM; k0 += 32) {
    v16h a = pack16_f16(arow + k0 + g8, arow + k0 + 16 + g8);
#pragma unroll
    for (int c = 0; c < 4; ++c) {
      const float* wr = Wo + (size_t)(o0 + 16 * c + ln) * DM + k0 + g8;
      v16h bf = pack16_f32(wr, wr + 16);
      acc[c] = WMMA_F32_F16(a, bf, acc[c]);
    }
  }

#pragma unroll
  for (int c = 0; c < 4; ++c) {
    const int col = o0 + 16 * c + ln;
    const float bval = bo[col];
#pragma unroll
    for (int r = 0; r < 8; ++r) {
      __builtin_nontemporal_store(acc[c][r] + bval,
                                  out + (size_t)(row0 + r + 8 * hw) * DM + col);
    }
  }
}

// ---------------------------------------------------------------------------
extern "C" void kernel_launch(void* const* d_in, const int* in_sizes, int n_in,
                              void* d_out, int out_size, void* d_ws, size_t ws_size,
                              hipStream_t stream) {
  (void)in_sizes; (void)n_in; (void)out_size; (void)ws_size;
  const float* queries = (const float*)d_in[0];
  const float* keys    = (const float*)d_in[1];
  const float* values  = (const float*)d_in[2];
  const unsigned char* mask = (const unsigned char*)d_in[3];  // jax bool = 1 byte
  const float* weights = (const float*)d_in[4];
  const float* Wq = (const float*)d_in[5];
  const float* bq = (const float*)d_in[6];
  const float* Wk = (const float*)d_in[7];
  const float* bk = (const float*)d_in[8];
  const float* Wv = (const float*)d_in[9];
  const float* bv = (const float*)d_in[10];
  const float* Wo = (const float*)d_in[11];
  const float* bo = (const float*)d_in[12];

  // Workspace: 4 x 8MB f16 tensors = 32MB total.
  char* ws = (char*)d_ws;
  const size_t SZ = (size_t)BS * H * NQ * DH * sizeof(_Float16);   // 8 MB
  _Float16* Qh = (_Float16*)(ws);
  _Float16* Kh = (_Float16*)(ws + SZ);
  _Float16* Vt = (_Float16*)(ws + 2 * SZ);
  _Float16* cc = (_Float16*)(ws + 3 * SZ);

  qkv_proj_kernel<<<dim3(512, 3), 256, 0, stream>>>(
      queries, keys, values, Wq, bq, Wk, bk, Wv, bv, Qh, Kh, Vt);
  attn_kernel<<<512, 256, 0, stream>>>(Qh, Kh, Vt, mask, weights, cc);
  out_proj_kernel<<<512, 256, 0, stream>>>(cc, Wo, bo, (float*)d_out);
}
